// KGNN_56899726737800
// MI455X (gfx1250) — compile-verified, matrix operand order checked by reading, not run on hardware
//
#include <hip/hip_runtime.h>
#include <hip/hip_bf16.h>
#include <math.h>

#define NNODES 50000
#define NEDGES 800000
#define INDIM 128
#define HIDDIM 64
#define OUTDIM 40
#define NREL 8

typedef __attribute__((ext_vector_type(16))) _Float16 v16h;
typedef __attribute__((ext_vector_type(8)))  _Float16 v8h;
typedef __attribute__((ext_vector_type(8)))  float    v8f;

// ---------------------------------------------------------------- convert x
__global__ void f32_to_f16_kernel(const float* __restrict__ in,
                                  _Float16* __restrict__ out, int n) {
  int i = blockIdx.x * blockDim.x + threadIdx.x;
  if (i < n) out[i] = (_Float16)in[i];
}

// --------------------------------------------------- pack weights for WMMA B
// Layout: Bp[r][kt][nt][lane][h] halves (512 per 32x16 tile), lane = column,
// K-half mapping identical to the 16-bit A-matrix VGPR layout so the GEMM
// kernel loads each lane's v16h as one contiguous 32B chunk.
__global__ void pack_weights_kernel(const float* __restrict__ W,     // [R][K][64]
                                    const float* __restrict__ Wroot, // [K][64]
                                    _Float16* __restrict__ Bp, int K, int R) {
  int idx = blockIdx.x * blockDim.x + threadIdx.x;
  int KT = K >> 5;
  int total = (R + 1) * KT * 4 * 512;
  if (idx >= total) return;
  int h  = idx & 15;
  int l  = (idx >> 4) & 31;
  int nt = (idx >> 9) & 3;
  int kt = (idx >> 11) % KT;
  int r  = (idx >> 11) / KT;
  int n  = nt * 16 + (l & 15);
  int kk = (h < 8) ? ((l >> 4) * 8 + h) : (16 + (l >> 4) * 8 + (h - 8));
  int k  = kt * 32 + kk;
  float v = (r < R) ? W[((size_t)r * K + k) * 64 + n]
                    : Wroot[(size_t)k * 64 + n];
  Bp[idx] = (_Float16)v;
}

// -------------------------------------------------------------- WMMA GEMM
// One wave computes a 16(M) x 64(N) tile for one relation r (blockIdx.y).
// r < R  : write H[r][row][col]   (per-relation features, gathered later)
// r == R : write agg[row][col] = root-term + bias (accumulator initializer)
template <int KTILES>
__global__ __launch_bounds__(256) void rgcn_gemm_wmma(
    const _Float16* __restrict__ Xh,   // [N, K] row-major f16
    const _Float16* __restrict__ Bp,   // [(R+1)][KTILES][4][512] f16 packed
    const float* __restrict__ bias,    // [64]
    float* __restrict__ H,             // [R][N][64]
    float* __restrict__ agg,           // [N][64]
    int nNodes, int R) {
  const int K    = KTILES * 32;
  const int lane = threadIdx.x & 31;
  const int wave = threadIdx.x >> 5;
  const int tileM = blockIdx.x * (blockDim.x >> 5) + wave;
  if (tileM * 16 >= nNodes) return;
  const int r = blockIdx.y;

  const int arow_i = tileM * 16 + (lane & 15);
  const int khalf  = lane >> 4;  // lanes 16..31 hold the upper K-halves

  v8f acc[4];
#pragma unroll
  for (int t = 0; t < 4; ++t) acc[t] = (v8f){};

  const _Float16* __restrict__ arow  = Xh + (size_t)arow_i * K;
  const _Float16* __restrict__ bbase = Bp + (size_t)r * KTILES * 4 * 512;

#pragma unroll
  for (int kt = 0; kt < KTILES; ++kt) {
    union { v16h v; v8h h[2]; } a;
    a.h[0] = *(const v8h*)(arow + kt * 32 + khalf * 8);        // K local 0..15
    a.h[1] = *(const v8h*)(arow + kt * 32 + 16 + khalf * 8);   // K local 16..31
    const _Float16* __restrict__ bkt = bbase + kt * (4 * 512);
#pragma unroll
    for (int nt = 0; nt < 4; ++nt) {
      v16h b = *(const v16h*)(bkt + nt * 512 + lane * 16);
      acc[nt] = __builtin_amdgcn_wmma_f32_16x16x32_f16(
          false, a.v, false, b, (short)0, acc[nt], false, false);
    }
  }

  // C/D layout: lanes 0-15 -> M = v, lanes 16-31 -> M = v + 8; N = lane & 15
  const int mb = tileM * 16 + ((lane & 16) ? 8 : 0);
#pragma unroll
  for (int nt = 0; nt < 4; ++nt) {
    const int col = nt * 16 + (lane & 15);
    if (r < R) {
      float* __restrict__ out = H + (size_t)r * nNodes * 64;
#pragma unroll
      for (int v = 0; v < 8; ++v)
        out[(size_t)(mb + v) * 64 + col] = acc[nt][v];
    } else {
      const float bv = bias[col];
#pragma unroll
      for (int v = 0; v < 8; ++v)
        agg[(size_t)(mb + v) * 64 + col] = acc[nt][v] + bv;
    }
  }
}

// ------------------------------------------------------------ edge counting
__global__ void edge_count_kernel(const int* __restrict__ etype,
                                  const int* __restrict__ dst,
                                  int* __restrict__ cnt, int E, int nNodes) {
  int e = blockIdx.x * blockDim.x + threadIdx.x;
  if (e < E) atomicAdd(&cnt[etype[e] * nNodes + dst[e]], 1);
}

__global__ void make_inv_kernel(const int* __restrict__ cnt,
                                float* __restrict__ inv, int n) {
  int i = blockIdx.x * blockDim.x + threadIdx.x;
  if (i < n) {
    int c = cnt[i];
    inv[i] = 1.0f / (float)(c > 0 ? c : 1);
  }
}

// ------------------------------------------------- mean-scatter over edges
// 64 threads per edge; each adds H[r][src][k] * inv(r,dst) into agg[dst][k].
__global__ __launch_bounds__(256) void edge_scatter_kernel(
    const int* __restrict__ etype, const int* __restrict__ src,
    const int* __restrict__ dst, const float* __restrict__ inv,
    const float* __restrict__ H, float* __restrict__ agg, int E, int nNodes) {
  long long gid = (long long)blockIdx.x * blockDim.x + threadIdx.x;
  int e = (int)(gid >> 6);
  int k = (int)(gid & 63);
  if (e >= E) return;
  int r = etype[e];
  int j = src[e];
  int i = dst[e];
  float s = inv[r * nNodes + i];
  float v = H[((size_t)r * nNodes + j) * 64 + k] * s;
  atomicAdd(&agg[(size_t)i * 64 + k], v);
}

// --------------------------------------------------- ReLU (+optional f16)
__global__ void finalize_relu_kernel(float* __restrict__ agg,
                                     _Float16* __restrict__ hf16, int n) {
  int i = blockIdx.x * blockDim.x + threadIdx.x;
  if (i >= n) return;
  float v = agg[i];
  v = v > 0.0f ? v : 0.0f;
  agg[i] = v;
  if (hf16) hf16[i] = (_Float16)v;
}

// -------------------------------------------- linear head + log_softmax
__global__ __launch_bounds__(256) void head_kernel(
    const float* __restrict__ h, const float* __restrict__ Wl,
    const float* __restrict__ bl, float* __restrict__ out, int nNodes) {
  __shared__ float sW[HIDDIM * OUTDIM];
  __shared__ float sb[OUTDIM];
  for (int t = threadIdx.x; t < HIDDIM * OUTDIM; t += blockDim.x) sW[t] = Wl[t];
  for (int t = threadIdx.x; t < OUTDIM; t += blockDim.x) sb[t] = bl[t];
  __syncthreads();
  int node = blockIdx.x * blockDim.x + threadIdx.x;
  if (node >= nNodes) return;

  float hv[HIDDIM];
#pragma unroll
  for (int k = 0; k < HIDDIM; ++k) hv[k] = h[(size_t)node * HIDDIM + k];

  float logit[OUTDIM];
  float mx = -3.4e38f;
#pragma unroll 4
  for (int o = 0; o < OUTDIM; ++o) {
    float s = sb[o];
#pragma unroll
    for (int k = 0; k < HIDDIM; ++k) s = fmaf(hv[k], sW[k * OUTDIM + o], s);
    logit[o] = s;
    mx = fmaxf(mx, s);
  }
  float se = 0.0f;
#pragma unroll
  for (int o = 0; o < OUTDIM; ++o) se += __expf(logit[o] - mx);
  float lse = mx + __logf(se);
#pragma unroll
  for (int o = 0; o < OUTDIM; ++o)
    out[(size_t)node * OUTDIM + o] = logit[o] - lse;
}

// ---------------------------------------------------------------- launcher
extern "C" void kernel_launch(void* const* d_in, const int* in_sizes, int n_in,
                              void* d_out, int out_size, void* d_ws,
                              size_t ws_size, hipStream_t stream) {
  (void)in_sizes; (void)n_in; (void)out_size; (void)ws_size;
  const float* x     = (const float*)d_in[0];
  const int*   eidx  = (const int*)d_in[1];   // [2, E]
  const int*   etype = (const int*)d_in[2];   // [E]
  const float* W1    = (const float*)d_in[3];
  const float* root1 = (const float*)d_in[4];
  const float* b1    = (const float*)d_in[5];
  const float* W2    = (const float*)d_in[6];
  const float* root2 = (const float*)d_in[7];
  const float* b2    = (const float*)d_in[8];
  const float* Wl    = (const float*)d_in[9];
  const float* bl    = (const float*)d_in[10];
  float* out = (float*)d_out;
  const int* src = eidx;
  const int* dst = eidx + NEDGES;

  // workspace carve-out (256B aligned)
  char* base = (char*)d_ws;
  size_t off = 0;
  auto carve = [&](size_t bytes) {
    void* p = base + off;
    off = (off + bytes + 255) & ~(size_t)255;
    return p;
  };
  _Float16* xh  = (_Float16*)carve((size_t)NNODES * INDIM * 2);
  _Float16* hh  = (_Float16*)carve((size_t)NNODES * HIDDIM * 2);
  _Float16* Bp1 = (_Float16*)carve((size_t)(NREL + 1) * (INDIM / 32) * 4 * 512 * 2);
  _Float16* Bp2 = (_Float16*)carve((size_t)(NREL + 1) * (HIDDIM / 32) * 4 * 512 * 2);
  float*    H   = (float*)carve((size_t)NREL * NNODES * 64 * 4);
  float*    agg = (float*)carve((size_t)NNODES * 64 * 4);
  int*      cnt = (int*)carve((size_t)NREL * NNODES * 4);
  float*    inv = (float*)carve((size_t)NREL * NNODES * 4);

  const int B = 256;
  const int mtiles = NNODES / 16;                    // 3125, exact
  dim3 gemm_grid((mtiles + 7) / 8, NREL + 1);

  // shared edge statistics (both layers use the same graph)
  hipMemsetAsync(cnt, 0, (size_t)NREL * NNODES * 4, stream);
  edge_count_kernel<<<(NEDGES + B - 1) / B, B, 0, stream>>>(etype, dst, cnt,
                                                            NEDGES, NNODES);
  make_inv_kernel<<<(NREL * NNODES + B - 1) / B, B, 0, stream>>>(
      cnt, inv, NREL * NNODES);

  // precision conversion + weight packing
  f32_to_f16_kernel<<<(NNODES * INDIM + B - 1) / B, B, 0, stream>>>(
      x, xh, NNODES * INDIM);
  {
    int tot1 = (NREL + 1) * (INDIM / 32) * 4 * 512;
    pack_weights_kernel<<<(tot1 + B - 1) / B, B, 0, stream>>>(W1, root1, Bp1,
                                                              INDIM, NREL);
    int tot2 = (NREL + 1) * (HIDDIM / 32) * 4 * 512;
    pack_weights_kernel<<<(tot2 + B - 1) / B, B, 0, stream>>>(W2, root2, Bp2,
                                                              HIDDIM, NREL);
  }

  // ---- layer 1: H[r] = x @ W1[r]; agg = x @ root1 + b1
  rgcn_gemm_wmma<INDIM / 32><<<gemm_grid, B, 0, stream>>>(xh, Bp1, b1, H, agg,
                                                          NNODES, NREL);
  {
    long long tot = (long long)NEDGES * 64;
    edge_scatter_kernel<<<(unsigned)((tot + B - 1) / B), B, 0, stream>>>(
        etype, src, dst, inv, H, agg, NEDGES, NNODES);
  }
  finalize_relu_kernel<<<(NNODES * HIDDIM + B - 1) / B, B, 0, stream>>>(
      agg, hh, NNODES * HIDDIM);

  // ---- layer 2: H[r] = h @ W2[r]; agg = h @ root2 + b2
  rgcn_gemm_wmma<HIDDIM / 32><<<gemm_grid, B, 0, stream>>>(hh, Bp2, b2, H, agg,
                                                           NNODES, NREL);
  {
    long long tot = (long long)NEDGES * 64;
    edge_scatter_kernel<<<(unsigned)((tot + B - 1) / B), B, 0, stream>>>(
        etype, src, dst, inv, H, agg, NEDGES, NNODES);
  }
  finalize_relu_kernel<<<(NNODES * HIDDIM + B - 1) / B, B, 0, stream>>>(
      agg, (_Float16*)nullptr, NNODES * HIDDIM);

  // ---- head: logits = h2 @ Wl + bl; log_softmax
  head_kernel<<<(NNODES + B - 1) / B, B, 0, stream>>>(agg, Wl, bl, out, NNODES);
}
